// SuperOper2D_78898549228086
// MI455X (gfx1250) — matrix operand order repaired and checked by reading, not compiled, not run on hardware
//
#include <hip/hip_runtime.h>
#include <math.h>

typedef __attribute__((ext_vector_type(16))) _Float16 v16h;
typedef __attribute__((ext_vector_type(8)))  _Float16 h8;
typedef __attribute__((ext_vector_type(8)))  float    v8f;
typedef __attribute__((ext_vector_type(4)))  float    f4;

#define IMGH 512
#define IMGW 512
#define NIMG 8
#define CI   8
#define CO   32
// im2col reduction: 9 taps * 16 chan (x | x^2) = 144, padded to 160 = 5 chunks of 32

__device__ __forceinline__ v16h cat16(h8 lo, h8 hi) {
  v16h r;
#pragma unroll
  for (int q = 0; q < 8; ++q) { r[q] = lo[q]; r[q + 8] = hi[q]; }
  return r;
}

__device__ __forceinline__ int iclamp(int v, int lo, int hi) {
  return v < lo ? lo : (v > hi ? hi : v);
}

// ---------------------------------------------------------------------------
// Pass 1: dual conv (x*W1 + x^2*W2 + biases) as implicit GEMM on WMMA f16.
// Block = 256 threads (8 wave32 waves) = 128 pixels of one image row.
// Each wave: M=16 pixels, N=32 channels (2 tiles), K=160 (5x v_wmma 16x16x32).
// Epilogue transposes D fragments through LDS so the f16 feature map is
// written channel-planar (NCHW) with coalesced global_store_b128.
// Feature map is f16: 128 MB total -> resident in the 192 MB L2.
// ---------------------------------------------------------------------------
__global__ __launch_bounds__(256) void conv_dual_wmma_kernel(
    const float* __restrict__ x, const float* __restrict__ W1,
    const float* __restrict__ b1, const float* __restrict__ W2,
    const float* __restrict__ b2, _Float16* __restrict__ featp) {
  // in_lds: [4 rows][132 cols][16 ch f16]; row 3 is a permanent zero row for
  // the K=144..159 padding taps. Cols cover jbase-1 .. jbase+128 (+2 slack).
  // Reused after the MMA loop as a [32 ch][136] f16 transpose buffer.
  __shared__ __align__(16) _Float16 in_lds[4 * 132 * 16];   // 16896 B
  // w_lds: [32 n][160 k] f16, k = tap*16 + c (c<8 -> W1, c>=8 -> W2), padded.
  __shared__ __align__(16) _Float16 w_lds[32 * 160];

  const int tid   = threadIdx.x;
  const int bw    = blockIdx.x & 3;          // W / 128 = 4 tiles per row
  const int ri    = blockIdx.x >> 2;
  const int row   = ri & (IMGH - 1);
  const int img   = ri >> 9;
  const int jbase = bw * 128;

  // ---- stage input tile as f16 {x, x^2} pairs (SAME zero padding) ----
  for (int e = tid; e < 4 * 132; e += 256) {
    const int r  = e / 132;
    const int cc = e - r * 132;
    const int y  = row + r - 1;
    const int j  = jbase + cc - 1;
    h8 lo, hi;
#pragma unroll
    for (int q = 0; q < 8; ++q) { lo[q] = (_Float16)0.f; hi[q] = (_Float16)0.f; }
    if (r < 3 && y >= 0 && y < IMGH && cc < 130 && j >= 0 && j < IMGW) {
      const f4* p = reinterpret_cast<const f4*>(
          x + (((size_t)img * IMGH + y) * IMGW + j) * CI);
      f4 v0 = p[0], v1 = p[1];
#pragma unroll
      for (int q = 0; q < 4; ++q) {
        lo[q]     = (_Float16)v0[q];
        lo[q + 4] = (_Float16)v1[q];
        hi[q]     = (_Float16)(v0[q] * v0[q]);
        hi[q + 4] = (_Float16)(v1[q] * v1[q]);
      }
    }
    *reinterpret_cast<h8*>(&in_lds[(r * 132 + cc) * 16])     = lo;
    *reinterpret_cast<h8*>(&in_lds[(r * 132 + cc) * 16 + 8]) = hi;
  }

  // ---- stage im2col weight matrix [n][160] f16 ----
  for (int e = tid; e < 32 * 160; e += 256) {
    const int n = e / 160;
    const int k = e - n * 160;
    _Float16 v = (_Float16)0.f;
    if (k < 144) {
      const int tap = k >> 4;
      const int c   = k & 15;
      const float* src = (c < 8) ? W1 : W2;
      v = (_Float16)src[((size_t)tap * CI + (c & 7)) * CO + n];  // (ky,kx,ci,co)
    }
    w_lds[e] = v;
  }
  __syncthreads();

  const int wave = tid >> 5;   // wave32
  const int lane = tid & 31;
  const int m    = lane & 15;  // A-matrix row (pixel) and D-matrix column (n)
  const int half = lane >> 4;  // K-half select per ISA 16-bit A/B layouts

  v8f acc0, acc1;
  const float bias0 = b1[m] + b2[m];
  const float bias1 = b1[m + 16] + b2[m + 16];
#pragma unroll
  for (int r = 0; r < 8; ++r) { acc0[r] = bias0; acc1[r] = bias1; }

  const h8* inv = reinterpret_cast<const h8*>(in_lds);
  const h8* wv  = reinterpret_cast<const h8*>(w_lds);
  const int colm = wave * 16 + m;   // LDS col for this lane's pixel

#pragma unroll
  for (int kc = 0; kc < 5; ++kc) {
    const int tA = 2 * kc, tB = 2 * kc + 1;      // two 16-wide taps per chunk
    const int rA = tA / 3, dA = tA - rA * 3;     // tap -> (lds row, kx)
    const int rB = tB / 3, dB = tB - rB * 3;     // tap 9 -> zero row 3
    // A fragment: lane m, K = {0..7,16..23} (half 0) / {8..15,24..31} (half 1)
    h8 aLo = inv[(rA * 132 + colm + dA) * 2 + half];
    h8 aHi = inv[(rB * 132 + colm + dB) * 2 + half];
    v16h a = cat16(aLo, aHi);
    // B fragments: column n = m (+16), K = {0..7,16..23} (half 0) / {8..15,24..31} (half 1)
    const int kb = kc * 4 + half;                // in h8 units (row = 20 h8)
    v16h bt0 = cat16(wv[m * 20 + kb],        wv[m * 20 + kb + 2]);
    v16h bt1 = cat16(wv[(m + 16) * 20 + kb], wv[(m + 16) * 20 + kb + 2]);
    acc0 = __builtin_amdgcn_wmma_f32_16x16x32_f16(false, a, false, bt0,
                                                  (short)0, acc0, false, false);
    acc1 = __builtin_amdgcn_wmma_f32_16x16x32_f16(false, a, false, bt1,
                                                  (short)0, acc1, false, false);
  }

  // ---- epilogue: transpose D fragments through LDS -> planar f16 stores ----
  __syncthreads();                 // all waves done reading in_lds
  _Float16* tr = in_lds;           // reused as [32 ch][136] (4352 halfs)
  // D layout: lane holds N = lane%16 (+tile*16), M = r + half*8 across VGPRs
#pragma unroll
  for (int r = 0; r < 8; ++r) {
    const int px = wave * 16 + r + half * 8;     // pixel within the 128-tile
    tr[m * 136 + px]        = (_Float16)acc0[r];
    tr[(m + 16) * 136 + px] = (_Float16)acc1[r];
  }
  __syncthreads();
  // 32 channels x 128 px f16 = 512 chunks of 16 B; 2 chunks per thread
  for (int L = tid; L < 512; L += 256) {
    const int c    = L >> 4;
    const int part = L & 15;
    h8 v = *reinterpret_cast<const h8*>(&tr[c * 136 + part * 8]);
    *reinterpret_cast<h8*>(
        &featp[(((size_t)img * CO + c) * IMGH + row) * IMGW + jbase + part * 8]) = v;
  }
}

// ---------------------------------------------------------------------------
// Pass 2: per-channel constant bilinear translation. floor(px) = j + floor(sx)
// with kx = floor(sx) constant per channel, so each (channel, output row
// segment of 64) needs exactly two source rows -> 72 halfs of each staged in LDS with
// coalesced loads from the planar f16 feature map, blended, and stored as
// coalesced 128 B f32 groups (32 consecutive channels per 32 lanes).
// ---------------------------------------------------------------------------
__global__ __launch_bounds__(256) void shift_bilinear_kernel(
    const _Float16* __restrict__ featp, const float* __restrict__ txy,
    float* __restrict__ out) {
  __shared__ __align__(16) _Float16 tile[32 * 2 * 74];  // per-ch 2 rows, padded
  __shared__ float s_sx[32], s_wy0[32], s_wy1[32];
  __shared__ int   s_kx[32], s_y0[32], s_y1[32];

  const int tid = threadIdx.x;
  const int jt  = blockIdx.x & 7;            // W / 64 = 8 segments per row
  const int ri  = blockIdx.x >> 3;
  const int i   = ri & (IMGH - 1);
  const int b   = ri >> 9;
  const int j0  = jt * 64;

  // per-channel shift constants (one thread per channel)
  if (tid < 32) {
    const float sx = txy[tid * 2 + 0] * ((IMGW - 1) * 0.5f);
    const float sy = txy[tid * 2 + 1] * ((IMGH - 1) * 0.5f);
    const int   kx = (int)floorf(sx);
    const int   ky = (int)floorf(sy);
    const int   y0 = iclamp(i + ky, 0, IMGH - 1);
    const int   y1 = iclamp(i + ky + 1, 0, IMGH - 1);
    const float py = (float)i + sy;
    s_sx[tid]  = sx;
    s_kx[tid]  = kx;
    s_y0[tid]  = y0;
    s_y1[tid]  = y1;
    s_wy0[tid] = (float)y1 - py;   // weights from CLIPPED coords (as reference)
    s_wy1[tid] = py - (float)y0;
  }
  __syncthreads();

  // stage: per channel, 2 source rows x 72 halfs (edge-replicated via clamp)
  for (int e = tid; e < 32 * 2 * 72; e += 256) {
    const int c  = e / 144;
    const int q  = e - c * 144;
    const int rr = q / 72;
    const int xx = q - rr * 72;
    const int yrow = rr ? s_y1[c] : s_y0[c];
    const int xsrc = iclamp(j0 + s_kx[c] + xx, 0, IMGW - 1);
    tile[(c * 2 + rr) * 74 + xx] =
        featp[(((size_t)b * CO + c) * IMGH + yrow) * IMGW + xsrc];
  }
  __syncthreads();

  const int c      = tid & 31;
  const int jjbase = (tid >> 5) * 8;
  const float sx  = s_sx[c];
  const int   kx  = s_kx[c];
  const float wy0 = s_wy0[c], wy1 = s_wy1[c];
  const int   xs  = j0 + kx;                 // unclamped tile origin
  const _Float16* t0 = &tile[(c * 2 + 0) * 74];
  const _Float16* t1 = &tile[(c * 2 + 1) * 74];
  const size_t obase = (((size_t)b * IMGH + i) * IMGW + j0 + jjbase) * CO + c;

#pragma unroll
  for (int jj = 0; jj < 8; ++jj) {
    const int   j   = j0 + jjbase + jj;
    const float px  = (float)j + sx;
    const int   x0c = iclamp(j + kx, 0, IMGW - 1);
    const int   x1c = iclamp(j + kx + 1, 0, IMGW - 1);
    const float wx0 = (float)x1c - px;       // clipped-coord weights
    const float wx1 = px - (float)x0c;
    const float g00 = (float)t0[x0c - xs];
    const float g01 = (float)t0[x1c - xs];
    const float g10 = (float)t1[x0c - xs];
    const float g11 = (float)t1[x1c - xs];
    out[obase + (size_t)jj * CO] =
        wy0 * (wx0 * g00 + wx1 * g01) + wy1 * (wx0 * g10 + wx1 * g11);
  }
}

extern "C" void kernel_launch(void* const* d_in, const int* in_sizes, int n_in,
                              void* d_out, int out_size, void* d_ws, size_t ws_size,
                              hipStream_t stream) {
  const float* x   = (const float*)d_in[0];
  const float* W1  = (const float*)d_in[1];
  const float* b1  = (const float*)d_in[2];
  const float* W2  = (const float*)d_in[3];
  const float* b2  = (const float*)d_in[4];
  const float* txy = (const float*)d_in[5];

  _Float16* featp = (_Float16*)d_ws;  // planar NCHW f16: 8*32*512*512*2 = 128 MB
  float* out = (float*)d_out;

  // Pass 1: 8 imgs * 512 rows * 4 (128-pixel tiles) = 16384 blocks
  conv_dual_wmma_kernel<<<dim3(NIMG * IMGH * (IMGW / 128)), dim3(256), 0, stream>>>(
      x, W1, b1, W2, b2, featp);
  // Pass 2: 8 imgs * 512 rows * 8 (64-pixel segments) = 32768 blocks
  shift_bilinear_kernel<<<dim3(NIMG * IMGH * (IMGW / 64)), dim3(256), 0, stream>>>(
      featp, txy, out);
}
